// MultiModalAttention_28673201668268
// MI455X (gfx1250) — matrix operand order repaired and checked by reading, not compiled, not run on hardware
//
#include <hip/hip_runtime.h>
#include <hip/hip_bf16.h>
#include <cstdint>
#include <cstddef>

// ---------------- problem constants ----------------
constexpr int B_   = 2;
constexpr int S_   = 2048;
constexpr int HID_ = 2048;
constexpr int H_   = 8;
constexpr int D_   = 256;
// KVH = 1, GROUPS = 8
constexpr float ATTN_CLAMP = 50.0f;
constexpr float LN_THETA   = 9.210340371976184f; // ln(10000)

typedef __attribute__((ext_vector_type(16))) _Float16 v16h;
typedef __attribute__((ext_vector_type(8)))  float    v8f;
typedef __attribute__((ext_vector_type(4)))  unsigned int v4u;
typedef __attribute__((ext_vector_type(8)))  int      v8i;
typedef __attribute__((ext_vector_type(4)))  int      v4i;
typedef _Float16 half_t;

// ---------------- WMMA helpers ----------------
__device__ __forceinline__ v8f vzero8() {
  v8f z;
#pragma unroll
  for (int i = 0; i < 8; ++i) z[i] = 0.0f;
  return z;
}

__device__ __forceinline__ v8f wmma32(v16h a, v16h b, v8f c) {
  // D = A(16x32 f16) * B(32x16 f16) + C(16x16 f32)
  return __builtin_amdgcn_wmma_f32_16x16x32_f16(
      /*neg_a=*/false, a, /*neg_b=*/false, b,
      /*c_mod=*/(short)0, c, /*reuse_a=*/false, /*reuse_b=*/false);
}

// A fragment 16x32 f16: element (m, k) at base[m*stride + k].
// Per lane the 16 elements form two contiguous 16B chunks -> 2x ds_load_b128.
__device__ __forceinline__ v16h load_a_frag(const half_t* base, int stride, int lane) {
  int row = lane & 15;
  int hl  = lane >> 4;
  v16h a;
#pragma unroll
  for (int v = 0; v < 8; ++v) {
    int k = ((v & 3) << 1) + (hl << 3) + ((v >> 2) << 4);
    a[2 * v]     = base[row * stride + k];
    a[2 * v + 1] = base[row * stride + k + 1];
  }
  return a;
}

// B fragment 32x16 f16, element (k, n) stored transposed: base[n*stride + k].
__device__ __forceinline__ v16h load_b_t(const half_t* base, int stride, int lane) {
  int n    = lane & 15;
  int koff = (lane >> 4) << 3;
  v16h b;
#pragma unroll
  for (int v = 0; v < 8; ++v) {
    int k = koff + ((v & 3) << 1) + ((v >> 2) << 4);
    b[2 * v]     = base[n * stride + k];
    b[2 * v + 1] = base[n * stride + k + 1];
  }
  return b;
}

// ---------------- CDNA5 async global->LDS (ASYNCcnt-tracked) ----------------
__device__ __forceinline__ void async_load_b128(uint32_t lds_off, const half_t* gptr) {
  asm volatile("global_load_async_to_lds_b128 %0, %1, off"
               :: "v"(lds_off), "v"((uint64_t)(uintptr_t)gptr)
               : "memory");
}
__device__ __forceinline__ void wait_async0() {
  asm volatile("s_wait_asynccnt 0" ::: "memory");
}

// ---------------- CDNA5 Tensor Data Mover: 2D f16 tile -> LDS ----------------
// D# per ISA ch8: group0 = {count|flags, lds_addr, global_addr[31:0],
// global_addr[56:32]|type<<30}; group1 packs data_size/pad/tensor/tile dims.
// pad_interval code c -> 2^(c+1) DWORDs, pad_amount code a -> (a+1) DWORDs.
__device__ __forceinline__ void tdm_load_2d_f16(
    uint32_t lds_off, const void* gp,
    uint32_t tile0, uint32_t tile1,       // tile dims (elements), dim0 contiguous
    uint32_t ten0,  uint32_t ten1,        // tensor dims for OOB (elements)
    uint32_t stride0,                     // dim0 stride (elements)
    uint32_t pic, uint32_t pac) {         // pad interval / amount codes
  uint64_t ga = (uint64_t)(uintptr_t)gp;
  v4u g0;
  g0[0] = 1u;                                    // count=1 (valid user D#)
  g0[1] = lds_off;                               // lds_addr (bytes)
  g0[2] = (uint32_t)ga;                          // global_addr[31:0]
  g0[3] = (uint32_t)((ga >> 32) & 0x01ffffffu) | (2u << 30); // addr hi | type=2
  v8i g1;
  g1[0] = (int)((1u << 16)            // data_size = 2 bytes
              | (1u << 20)            // pad_enable
              | (pic << 22)           // pad_interval
              | (pac << 25));         // pad_amount
  g1[1] = (int)((ten0 & 0xffffu) << 16);               // tensor_dim0[15:0]
  g1[2] = (int)((ten0 >> 16) | ((ten1 & 0xffffu) << 16)); // dim0 hi | dim1 lo
  g1[3] = (int)((ten1 >> 16) | (tile0 << 16));         // dim1 hi | tile_dim0
  g1[4] = (int)(tile1 & 0xffffu);                      // tile_dim1 (tile_dim2=0)
  g1[5] = (int)stride0;                                // tensor_dim0_stride lo
  g1[6] = 0;
  g1[7] = 0;
  v4i g2 = {0, 0, 0, 0};
  v4i g3 = {0, 0, 0, 0};
#if defined(__clang_major__) && (__clang_major__ >= 23)
  v8i gz = {0, 0, 0, 0, 0, 0, 0, 0};
  __builtin_amdgcn_tensor_load_to_lds(g0, g1, g2, g3, gz, 0);
#else
  __builtin_amdgcn_tensor_load_to_lds(g0, g1, g2, g3, 0);
#endif
}

// ---------------- f32 -> f16 convert ----------------
__global__ void f32_to_f16_kernel(const float* __restrict__ in,
                                  half_t* __restrict__ out, size_t n) {
  size_t i  = (size_t)blockIdx.x * blockDim.x + threadIdx.x;
  size_t st = (size_t)gridDim.x * blockDim.x;
  for (; i < n; i += st) out[i] = (half_t)in[i];
}

// ---------------- transpose + convert: out(N,K) f16 = in(K,N) f32 ^T ----------
__global__ __launch_bounds__(256) void transpose_w_kernel(
    const float* __restrict__ in, half_t* __restrict__ out, int K, int N) {
  __shared__ float tile[32][33];
  int k0 = blockIdx.y * 32;
  int n0 = blockIdx.x * 32;
  int tx = threadIdx.x & 31;
  int ty = threadIdx.x >> 5; // 0..7
#pragma unroll
  for (int j = 0; j < 32; j += 8)
    tile[ty + j][tx] = in[(size_t)(k0 + ty + j) * N + n0 + tx];
  __syncthreads();
#pragma unroll
  for (int j = 0; j < 32; j += 8)
    out[(size_t)(n0 + ty + j) * K + k0 + tx] = (half_t)tile[tx][ty + j];
}

// V: (B,S,D) f32 -> (B,D,S) f16 (transposed for PV B-fragments)
__global__ __launch_bounds__(256) void transpose_v_kernel(
    const float* __restrict__ in, half_t* __restrict__ out) {
  __shared__ float tile[32][33];
  int b  = blockIdx.z;
  int s0 = blockIdx.y * 32;
  int d0 = blockIdx.x * 32;
  int tx = threadIdx.x & 31;
  int ty = threadIdx.x >> 5;
#pragma unroll
  for (int j = 0; j < 32; j += 8)
    tile[ty + j][tx] = in[((size_t)b * S_ + s0 + ty + j) * D_ + d0 + tx];
  __syncthreads();
#pragma unroll
  for (int j = 0; j < 32; j += 8)
    out[((size_t)b * D_ + d0 + ty + j) * S_ + s0 + tx] = (half_t)tile[tx][ty + j];
}

// ---- tiled WMMA GEMM, double-buffered async staging ------------------------
// C = A(f16,MxK) * Bt(f16,NxK)^T + bias. 8 waves x 32 M-rows, 128 N-cols.
constexpr int BM  = 256;
constexpr int BN  = 128;
constexpr int BK  = 32;
constexpr int TST = 40;  // LDS tile row stride (f16): 80B rows, 16B-aligned quads

__global__ __launch_bounds__(256) void gemm_f16_kernel(
    const half_t* __restrict__ A, const half_t* __restrict__ Bt,
    const float* __restrict__ bias, float* __restrict__ C,
    int M, int N, int K) {
  __shared__ __align__(16) half_t As[2][BM * TST]; // 2 x 20480B
  __shared__ __align__(16) half_t Bs[2][BN * TST]; // 2 x 10240B

  const int t    = threadIdx.x;
  const int wave = t >> 5;
  const int lane = t & 31;
  const int m0   = blockIdx.y * BM;
  const int n0   = blockIdx.x * BN;

  auto stage = [&](int buf, int k0) {
    // A: 256 rows x 4 quads (1024), B: 128 rows x 4 quads (512) -> 6/thread
#pragma unroll
    for (int j = 0; j < 4; ++j) {
      int i   = t + j * 256;
      int row = i >> 2;
      int qc  = (i & 3) << 3;
      async_load_b128((uint32_t)(uintptr_t)&As[buf][row * TST + qc],
                      &A[(size_t)(m0 + row) * K + k0 + qc]);
    }
#pragma unroll
    for (int j = 0; j < 2; ++j) {
      int i   = t + j * 256;
      int row = i >> 2;
      int qc  = (i & 3) << 3;
      async_load_b128((uint32_t)(uintptr_t)&Bs[buf][row * TST + qc],
                      &Bt[(size_t)(n0 + row) * K + k0 + qc]);
    }
  };

  v8f acc0[8], acc1[8];
#pragma unroll
  for (int i = 0; i < 8; ++i) { acc0[i] = vzero8(); acc1[i] = vzero8(); }

  stage(0, 0);
  wait_async0();
  __syncthreads();

  int cur = 0;
  for (int k0 = 0; k0 < K; k0 += BK) {
    if (k0 + BK < K) stage(cur ^ 1, k0 + BK); // overlap next stage with compute

    v16h a0 = load_a_frag(&As[cur][(wave * 32) * TST], TST, lane);
    v16h a1 = load_a_frag(&As[cur][(wave * 32 + 16) * TST], TST, lane);
#pragma unroll
    for (int nt = 0; nt < 8; ++nt) {
      v16h b   = load_b_t(&Bs[cur][(nt * 16) * TST], TST, lane);
      acc0[nt] = wmma32(a0, b, acc0[nt]);
      acc1[nt] = wmma32(a1, b, acc1[nt]);
    }

    wait_async0();
    __syncthreads();
    cur ^= 1;
  }

  const int hl = lane >> 4;
  const int n  = lane & 15;
#pragma unroll
  for (int nt = 0; nt < 8; ++nt) {
    int   col = n0 + nt * 16 + n;
    float bv  = bias[col];
#pragma unroll
    for (int r = 0; r < 8; ++r) {
      int row0 = m0 + wave * 32 + r + 8 * hl;
      C[(size_t)row0 * N + col]        = acc0[nt][r] + bv;
      C[(size_t)(row0 + 16) * N + col] = acc1[nt][r] + bv;
    }
  }
}

// ---------------- RoPE ----------------
__global__ void rope_q_kernel(const float* __restrict__ Qpre,
                              const int* __restrict__ pos,
                              half_t* __restrict__ Qf) {
  int idx = blockIdx.x * blockDim.x + threadIdx.x; // B*S*H*(D/2) threads
  int d   = idx & (D_ / 2 - 1);
  int rem = idx >> 7;
  int h   = rem & (H_ - 1);
  rem >>= 3;
  int s = rem & (S_ - 1);
  int b = rem >> 11;

  size_t ib = (size_t)(b * S_ + s) * (H_ * D_) + h * D_ + d;
  float  x1 = Qpre[ib];
  float  x2 = Qpre[ib + D_ / 2];
  float  p  = (float)pos[b * S_ + s];
  float  f  = p * __expf(-((float)(2 * d) / (float)D_) * LN_THETA);
  float  c  = __cosf(f), sn = __sinf(f);

  size_t ob       = ((size_t)(b * H_ + h) * S_ + s) * D_ + d;
  Qf[ob]          = (half_t)(x1 * c - x2 * sn);
  Qf[ob + D_ / 2] = (half_t)(x2 * c + x1 * sn);
}

__global__ void rope_k_kernel(const float* __restrict__ Kpre,
                              const int* __restrict__ pos,
                              float* __restrict__ Kf32,
                              half_t* __restrict__ Kf16) {
  int idx = blockIdx.x * blockDim.x + threadIdx.x; // B*S*(D/2) threads
  int d   = idx & (D_ / 2 - 1);
  int rem = idx >> 7;
  int s   = rem & (S_ - 1);
  int b   = rem >> 11;

  size_t ib = (size_t)(b * S_ + s) * D_ + d;
  float  x1 = Kpre[ib];
  float  x2 = Kpre[ib + D_ / 2];
  float  p  = (float)pos[b * S_ + s];
  float  f  = p * __expf(-((float)(2 * d) / (float)D_) * LN_THETA);
  float  c  = __cosf(f), sn = __sinf(f);

  float o1 = x1 * c - x2 * sn;
  float o2 = x2 * c + x1 * sn;
  Kf32[ib]          = o1;
  Kf32[ib + D_ / 2] = o2;
  Kf16[ib]          = (half_t)o1;
  Kf16[ib + D_ / 2] = (half_t)o2;
}

// ---------------- flash attention (online softmax, WMMA, TDM staging) --------
constexpr int KT  = 32;   // keys per tile
constexpr int KST = 264;  // K LDS row stride over d (f16): 128+4 DWORDs
constexpr int VST = 40;   // V^T LDS row stride over keys: 16+4 DWORDs
constexpr int PST = 40;   // P staging stride

__global__ __launch_bounds__(128) void attn_kernel(
    const half_t* __restrict__ Qf,   // (B,H,S,D)
    const half_t* __restrict__ Kf,   // (B,S,D)   (KVH==1)
    const half_t* __restrict__ Vt,   // (B,D,S)   transposed
    const float*  __restrict__ mask, // (B,1,S,S)
    half_t* __restrict__ Of) {       // (B,S,H*D)
  __shared__ __align__(16) half_t Klds[2][KT * KST];  // double-buffered K tile
  __shared__ __align__(16) half_t Vlds[D_ * VST];     // [d][key], pad 4 DW/row
  __shared__ __align__(16) half_t Plds[4 * 16 * PST];

  const int t    = threadIdx.x;
  const int wave = t >> 5;
  const int lane = t & 31;
  const int bh   = blockIdx.y;
  const int b    = bh / H_;
  const int h    = bh % H_;
  const int q0   = blockIdx.x * 64 + wave * 16;
  const int hl   = lane >> 4;
  const int nl   = lane & 15;
  const float scale = 0.0625f; // 1/sqrt(256)

  // Q fragments held in registers for the whole key loop
  const half_t* qbase = Qf + ((size_t)bh * S_ + q0) * D_;
  v16h qf[8];
#pragma unroll
  for (int dk = 0; dk < 8; ++dk) qf[dk] = load_a_frag(qbase + dk * 32, D_, lane);

  v8f acc[16];
#pragma unroll
  for (int i = 0; i < 16; ++i) acc[i] = vzero8();
  float mrun[8], lrun[8];
#pragma unroll
  for (int r = 0; r < 8; ++r) { mrun[r] = -1e30f; lrun[r] = 0.0f; }

  const half_t* kgb = Kf + (size_t)b * S_ * D_;
  const half_t* vgb = Vt + (size_t)b * D_ * S_;

  // prologue: stage first K and V tiles via TDM (waves 0/1)
  if (wave == 0) {
    tdm_load_2d_f16((uint32_t)(uintptr_t)&Klds[0][0], kgb,
                    D_, KT, D_, S_, D_, /*pic=*/6, /*pac=*/3);
  } else if (wave == 1) {
    tdm_load_2d_f16((uint32_t)(uintptr_t)&Vlds[0], vgb,
                    KT, D_, S_, D_, S_, /*pic=*/3, /*pac=*/3);
  }
  __builtin_amdgcn_s_wait_tensorcnt(0);
  __syncthreads();

  int cur = 0;
  for (int kt = 0; kt < S_; kt += KT) {
    const half_t* Kc = &Klds[cur][0];

    // S tiles: 16 q rows x 32 keys, reduction over D (8 WMMAs each half)
    v8f s1 = vzero8(), s2 = vzero8();
#pragma unroll
    for (int dk = 0; dk < 8; ++dk) {
      v16h b1 = load_b_t(&Kc[0 * KST + dk * 32], KST, lane);   // keys 0..15
      s1 = wmma32(qf[dk], b1, s1);
      v16h b2 = load_b_t(&Kc[16 * KST + dk * 32], KST, lane);  // keys 16..31
      s2 = wmma32(qf[dk], b2, s2);
    }

    // early-issue next K tile into the other buffer; overlaps softmax + PV
    if (wave == 0 && kt + KT < S_) {
      tdm_load_2d_f16((uint32_t)(uintptr_t)&Klds[cur ^ 1][0],
                      kgb + (size_t)(kt + KT) * D_,
                      D_, KT, D_, S_, D_, /*pic=*/6, /*pac=*/3);
    }
    if (kt + KT < S_) { // hint next mask rows (global_prefetch_b8)
      __builtin_prefetch(mask + ((size_t)b * S_ + q0 + (t & 15)) * S_ + kt + KT, 0, 1);
    }

    // scale, clamp, mask
    float pm[8];
#pragma unroll
    for (int r = 0; r < 8; ++r) {
      int rq = q0 + r + 8 * hl;
      const float* mrow = mask + ((size_t)b * S_ + rq) * S_ + kt;
      float v1 = fminf(fmaxf(s1[r] * scale, -ATTN_CLAMP), ATTN_CLAMP) + mrow[nl];
      float v2 = fminf(fmaxf(s2[r] * scale, -ATTN_CLAMP), ATTN_CLAMP) + mrow[16 + nl];
      s1[r] = v1; s2[r] = v2;
      pm[r] = fmaxf(v1, v2);
    }
#pragma unroll
    for (int r = 0; r < 8; ++r) {
#pragma unroll
      for (int off = 1; off < 16; off <<= 1)
        pm[r] = fmaxf(pm[r], __shfl_xor(pm[r], off, 32));
    }
    // online softmax update
#pragma unroll
    for (int r = 0; r < 8; ++r) {
      float mnew = fmaxf(mrun[r], pm[r]);
      float sc   = __expf(mrun[r] - mnew);
      mrun[r]    = mnew;
      float p1   = __expf(s1[r] - mnew);
      float p2   = __expf(s2[r] - mnew);
      s1[r] = p1; s2[r] = p2;
      float rs = p1 + p2;
#pragma unroll
      for (int off = 1; off < 16; off <<= 1) rs += __shfl_xor(rs, off, 32);
      lrun[r] = lrun[r] * sc + rs;
#pragma unroll
      for (int dt = 0; dt < 16; ++dt) acc[dt][r] *= sc;
    }

    // P: C-layout -> A-layout via per-wave LDS staging (same-wave LDS in-order)
    half_t* pb = &Plds[wave * 16 * PST];
#pragma unroll
    for (int r = 0; r < 8; ++r) {
      pb[(r + 8 * hl) * PST + nl]      = (half_t)s1[r];
      pb[(r + 8 * hl) * PST + 16 + nl] = (half_t)s2[r];
    }
    v16h pa = load_a_frag(pb, PST, lane);

    // out += P(16x32) * V(32x256) ; V^T fragment = contiguous keys per lane
#pragma unroll
    for (int dt = 0; dt < 16; ++dt) {
      v16h bv = load_b_t(&Vlds[(dt * 16) * VST], VST, lane);
      acc[dt] = wmma32(pa, bv, acc[dt]);
    }

    __syncthreads(); // all waves done reading Vlds (and old K buffer)
    if (wave == 1 && kt + KT < S_) {
      tdm_load_2d_f16((uint32_t)(uintptr_t)&Vlds[0],
                      vgb + (kt + KT),
                      KT, D_, S_, D_, S_, /*pic=*/3, /*pac=*/3);
    }
    __builtin_amdgcn_s_wait_tensorcnt(0);
    __syncthreads(); // publish next K (already in flight) and V tiles
    cur ^= 1;
  }

  // epilogue: divide by l, store f16 (B,S,H*D)
#pragma unroll
  for (int r = 0; r < 8; ++r) {
    int   rq  = q0 + r + 8 * hl;
    float inv = 1.0f / lrun[r];
    half_t* orow = Of + ((size_t)b * S_ + rq) * (H_ * D_) + h * D_;
#pragma unroll
    for (int dt = 0; dt < 16; ++dt)
      orow[dt * 16 + nl] = (half_t)(acc[dt][r] * inv);
  }
}

// ---------------- host orchestration ----------------
extern "C" void kernel_launch(void* const* d_in, const int* in_sizes, int n_in,
                              void* d_out, int out_size, void* d_ws, size_t ws_size,
                              hipStream_t stream) {
  (void)in_sizes; (void)n_in; (void)out_size; (void)ws_size;
  const float* hid  = (const float*)d_in[0];
  const float* mask = (const float*)d_in[1];
  const int*   pos  = (const int*)d_in[2];
  const float* Wq   = (const float*)d_in[3];
  const float* bq   = (const float*)d_in[4];
  const float* Wk   = (const float*)d_in[5];
  const float* bk   = (const float*)d_in[6];
  const float* Wv   = (const float*)d_in[7];
  const float* bv   = (const float*)d_in[8];
  const float* Wo   = (const float*)d_in[9];
  const float* bo   = (const float*)d_in[10];

  float* out  = (float*)d_out;                        // (B,S,HID)
  float* kout = out + (size_t)B_ * S_ * HID_;         // (B,1,S,D)
  float* vout = kout + (size_t)B_ * S_ * D_;          // (B,1,S,D)

  const size_t NX  = (size_t)B_ * S_ * HID_;   // 8388608
  const size_t NWQ = (size_t)HID_ * H_ * D_;   // 4194304
  const size_t NWK = (size_t)HID_ * D_;        // 524288
  const size_t NQP = (size_t)B_ * S_ * H_ * D_;
  const size_t NKP = (size_t)B_ * S_ * D_;

  char* w = (char*)d_ws;
  half_t* Xh   = (half_t*)w; w += NX  * sizeof(half_t);
  half_t* Wqt  = (half_t*)w; w += NWQ * sizeof(half_t); // (N=2048, K=2048)
  half_t* Wkt  = (half_t*)w; w += NWK * sizeof(half_t); // (N=256,  K=2048)
  half_t* Wvt  = (half_t*)w; w += NWK * sizeof(half_t);
  half_t* Wot  = (half_t*)w; w += NWQ * sizeof(half_t);
  float*  Qpre = (float*)w;  w += NQP * sizeof(float);
  float*  Kpre = (float*)w;  w += NKP * sizeof(float);
  half_t* Qh   = (half_t*)w; w += NQP * sizeof(half_t); // (B,H,S,D)
  half_t* Kh   = (half_t*)w; w += NKP * sizeof(half_t); // (B,S,D)
  half_t* Vth  = (half_t*)w; w += NKP * sizeof(half_t); // (B,D,S)
  half_t* Oh   = (half_t*)w; w += NX  * sizeof(half_t); // (B,S,H*D)

  const int CB = 4096;
  f32_to_f16_kernel<<<CB, 256, 0, stream>>>(hid, Xh, NX);
  transpose_w_kernel<<<dim3(HID_ / 32, HID_ / 32), 256, 0, stream>>>(Wq, Wqt, HID_, HID_);
  transpose_w_kernel<<<dim3(D_ / 32,   HID_ / 32), 256, 0, stream>>>(Wk, Wkt, HID_, D_);
  transpose_w_kernel<<<dim3(D_ / 32,   HID_ / 32), 256, 0, stream>>>(Wv, Wvt, HID_, D_);
  transpose_w_kernel<<<dim3(HID_ / 32, HID_ / 32), 256, 0, stream>>>(Wo, Wot, HID_, HID_);

  const int Mrows = B_ * S_; // 4096
  gemm_f16_kernel<<<dim3(HID_ / BN, Mrows / BM), 256, 0, stream>>>(
      Xh, Wqt, bq, Qpre, Mrows, H_ * D_, HID_);
  gemm_f16_kernel<<<dim3(D_ / BN, Mrows / BM), 256, 0, stream>>>(
      Xh, Wkt, bk, Kpre, Mrows, D_, HID_);
  // V projection straight into d_out's v section ((B,1,S,D) == (B*S,D))
  gemm_f16_kernel<<<dim3(D_ / BN, Mrows / BM), 256, 0, stream>>>(
      Xh, Wvt, bv, vout, Mrows, D_, HID_);

  rope_q_kernel<<<(B_ * S_ * H_ * (D_ / 2)) / 256, 256, 0, stream>>>(Qpre, pos, Qh);
  rope_k_kernel<<<(B_ * S_ * (D_ / 2)) / 256, 256, 0, stream>>>(Kpre, pos, kout, Kh);
  transpose_v_kernel<<<dim3(D_ / 32, S_ / 32, B_), 256, 0, stream>>>(vout, Vth);

  attn_kernel<<<dim3(S_ / 64, B_ * H_), 128, 0, stream>>>(Qh, Kh, Vth, mask, Oh);

  gemm_f16_kernel<<<dim3(HID_ / BN, Mrows / BM), 256, 0, stream>>>(
      Oh, Wot, bo, out, Mrows, HID_, HID_);
}
